// AttentionKernel_88562225643899
// MI455X (gfx1250) — matrix operand order repaired and checked by reading, not compiled, not run on hardware
//
#include <hip/hip_runtime.h>

// ---------- problem constants (from reference) ----------
#define BATCH 4
#define SEQ   1024
#define DMODEL 1024
#define NHEAD 16
#define HDIM  64          // DMODEL / NHEAD
#define NBH   (BATCH*NHEAD)

typedef __bf16 v16bf __attribute__((ext_vector_type(16)));
typedef float  v8f   __attribute__((ext_vector_type(8)));
typedef uint4  __attribute__((may_alias)) uint4_a;

#define WMMA_BF16(a,b,c) \
  __builtin_amdgcn_wmma_f32_16x16x32_bf16(false,(a),false,(b),(short)0,(c),false,false)

// float -> bf16 (round to nearest even), returned as raw u16
__device__ __forceinline__ unsigned short f2bf_raw(float f) {
  unsigned u = __builtin_bit_cast(unsigned, f);
  unsigned r = u + 0x7FFFu + ((u >> 16) & 1u);
  return (unsigned short)(r >> 16);
}

union V16U { v16bf v; uint4_a u[2]; };

// A-operand load (16-bit A 16x32 layout): per lane two 8-elem chunks,
// second chunk is +16 elements (+32 bytes). Caller passes base already
// offset by koffA (= 8 * (lane>=16)).
__device__ __forceinline__ v16bf load_a16(const unsigned short* base) {
  const uint4_a* p = reinterpret_cast<const uint4_a*>(base);
  V16U t; t.u[0] = p[0]; t.u[1] = p[2];
  return t.v;
}

// B-operand load (16-bit B 32x16 layout): per lane two 8-elem chunks,
// second chunk is +16 elements (same K order as A; base offset by koffA).
__device__ __forceinline__ v16bf load_b16(const unsigned short* base) {
  const uint4_a* p = reinterpret_cast<const uint4_a*>(base);
  V16U t; t.u[0] = p[0]; t.u[1] = p[2];
  return t.v;
}

// ------------------------------------------------------------------
// Preprocess: fp32 (B,S,D) -> bf16 workspace
//   Qb[bh][s][64]  (scaled by 1/sqrt(64))
//   Kb[bh][s][64]
//   Vt[bh][64][S]  (transposed per head for contiguous B-operand loads)
// ------------------------------------------------------------------
__global__ __launch_bounds__(256)
void fa_prep_kernel(const float* __restrict__ q,
                    const float* __restrict__ k,
                    const float* __restrict__ v,
                    unsigned short* __restrict__ qb,
                    unsigned short* __restrict__ kb,
                    unsigned short* __restrict__ vt) {
  int idx = blockIdx.x * 256 + threadIdx.x;      // [bh][s][d] flat
  int d = idx & (HDIM - 1);
  int s = (idx >> 6) & (SEQ - 1);
  int h = (idx >> 16) & (NHEAD - 1);
  int b = idx >> 20;
  size_t src = ((size_t)(b * SEQ + s)) * DMODEL + h * HDIM + d;
  size_t bh  = (size_t)b * NHEAD + h;
  qb[idx] = f2bf_raw(q[src] * 0.125f);           // 64^-0.5
  kb[idx] = f2bf_raw(k[src]);
  vt[(bh * HDIM + d) * SEQ + s] = f2bf_raw(v[src]);
}

// ------------------------------------------------------------------
// Flash attention: 4 waves/block, 32 q-rows (two 16-row tiles) per wave.
// K/V B-operands are loaded once per chunk and reused by both Q tiles.
// grid.x = NBH * (SEQ/128) = 512
// ------------------------------------------------------------------
__global__ __launch_bounds__(128)
void fa_attn_kernel(const unsigned short* __restrict__ qb,
                    const unsigned short* __restrict__ kb,
                    const unsigned short* __restrict__ vt,
                    float* __restrict__ out) {
  // per-wave, per-tile P staging: 16 rows x 32 cols bf16, row stride 40
  __shared__ __align__(16) unsigned short plds_all[4][2][16 * 40];

  const int lane = threadIdx.x & 31;
  const int wave = threadIdx.x >> 5;
  const int wg   = blockIdx.x;
  const int qblk = wg & (SEQ / 128 - 1);         // 8 q-blocks of 128 rows
  const int h    = (wg >> 3) & (NHEAD - 1);
  const int b    = wg >> 7;
  const int bh   = b * NHEAD + h;
  const int q0   = qblk * 128 + wave * 32;       // this wave's 32 q rows

  const int ln    = lane & 15;
  const int half  = lane >> 4;
  const int koffA = half * 8;                    // A-operand K chunk base
  const int koffB = half * 16;                   // B-operand K run base

  const unsigned short* Qb = qb + (size_t)bh * SEQ * HDIM;
  const unsigned short* Kb = kb + (size_t)bh * SEQ * HDIM;
  const unsigned short* Vt = vt + (size_t)bh * HDIM * SEQ;

  // Q tile A-operands (resident for the whole loop): 2 tiles x (hd 0..31 / 32..63)
  v16bf qa[2][2];
#pragma unroll
  for (int t = 0; t < 2; ++t) {
    const unsigned short* qrow = Qb + (size_t)(q0 + 16 * t + ln) * HDIM;
    qa[t][0] = load_a16(qrow + koffA);
    qa[t][1] = load_a16(qrow + 32 + koffA);
  }

  v8f acc[2][4];
  float mrow[2][8], lrow[2][8];
#pragma unroll
  for (int t = 0; t < 2; ++t)
#pragma unroll
    for (int r = 0; r < 8; ++r) {
      mrow[t][r] = -__builtin_inff();
      lrow[t][r] = 0.f;
      acc[t][0][r] = 0.f; acc[t][1][r] = 0.f; acc[t][2][r] = 0.f; acc[t][3][r] = 0.f;
    }

  for (int kc = 0; kc < q0 + 32; kc += 32) {
    // ---- K B-operands: key tiles [kc, kc+16) and [kc+16, kc+32) ----
    const unsigned short* krow0 = Kb + (size_t)(kc + ln) * HDIM;
    const unsigned short* krow1 = Kb + (size_t)(kc + 16 + ln) * HDIM;
    v16bf kb00 = load_b16(krow0 + koffA);        // hd 0..31
    v16bf kb01 = load_b16(krow0 + 32 + koffA);   // hd 32..63
    v16bf kb10 = load_b16(krow1 + koffA);
    v16bf kb11 = load_b16(krow1 + 32 + koffA);

    // prefetch next chunk's K rows (near-cache hint)
    __builtin_prefetch(krow0 + 32 * HDIM, 0, 3);
    __builtin_prefetch(krow1 + 32 * HDIM, 0, 3);

    // ---- scores: both Q tiles share the K operands ----
    v8f s[2][2];
#pragma unroll
    for (int t = 0; t < 2; ++t) {
      s[t][0] = v8f{};
      s[t][0] = WMMA_BF16(qa[t][0], kb00, s[t][0]);
      s[t][0] = WMMA_BF16(qa[t][1], kb01, s[t][0]);
      s[t][1] = v8f{};
      s[t][1] = WMMA_BF16(qa[t][0], kb10, s[t][1]);
      s[t][1] = WMMA_BF16(qa[t][1], kb11, s[t][1]);
    }

    // ---- causal mask near/past the diagonal (pure VALU, no WMMA inside) ----
#pragma unroll
    for (int t = 0; t < 2; ++t) {
      if (kc + 31 > q0 + 16 * t) {
#pragma unroll
        for (int r = 0; r < 8; ++r) {
          int row = q0 + 16 * t + r + 8 * half;
          if (kc + ln > row)      s[t][0][r] = -__builtin_inff();
          if (kc + 16 + ln > row) s[t][1][r] = -__builtin_inff();
        }
      }
    }

    // ---- online softmax per tile ----
#pragma unroll
    for (int t = 0; t < 2; ++t) {
      float rmax[8];
#pragma unroll
      for (int r = 0; r < 8; ++r) rmax[r] = fmaxf(s[t][0][r], s[t][1][r]);
#pragma unroll
      for (int m = 1; m < 16; m <<= 1)
#pragma unroll
        for (int r = 0; r < 8; ++r)
          rmax[r] = fmaxf(rmax[r], __shfl_xor(rmax[r], m, 32));

      float p0[8], p1[8], rsum[8], ascale[8];
#pragma unroll
      for (int r = 0; r < 8; ++r) {
        float mn  = fmaxf(mrow[t][r], rmax[r]);
        ascale[r] = __expf(mrow[t][r] - mn);
        mrow[t][r] = mn;
        p0[r]     = __expf(s[t][0][r] - mn);
        p1[r]     = __expf(s[t][1][r] - mn);
        rsum[r]   = p0[r] + p1[r];
      }
#pragma unroll
      for (int m = 1; m < 16; m <<= 1)
#pragma unroll
        for (int r = 0; r < 8; ++r)
          rsum[r] += __shfl_xor(rsum[r], m, 32);
#pragma unroll
      for (int r = 0; r < 8; ++r) {
        lrow[t][r] = lrow[t][r] * ascale[r] + rsum[r];
        acc[t][0][r] *= ascale[r];
        acc[t][1][r] *= ascale[r];
        acc[t][2][r] *= ascale[r];
        acc[t][3][r] *= ascale[r];
      }

      // P (C-layout) -> LDS (bf16)
      unsigned short* plds = plds_all[wave][t];
#pragma unroll
      for (int r = 0; r < 8; ++r) {
        int row = r + 8 * half;
        plds[row * 40 + ln]      = f2bf_raw(p0[r]);
        plds[row * 40 + 16 + ln] = f2bf_raw(p1[r]);
      }
    }

    // ---- V B-operands: load ALL FOUR before any PV WMMA so the loads can
    // be claused and waits staggered (avoid register-reuse serialization) ----
    v16bf vb[4];
#pragma unroll
    for (int j = 0; j < 4; ++j) {
      const unsigned short* vrow = Vt + (size_t)(j * 16 + ln) * SEQ + kc;
      vb[j] = load_b16(vrow + koffA);
      __builtin_prefetch(vrow + 32, 0, 3);       // next chunk's V keys
    }

    // P A-operands (same-wave DS ops are in-order; compiler inserts waits)
    v16bf pa[2];
    pa[0] = load_a16(plds_all[wave][0] + ln * 40 + koffA);
    pa[1] = load_a16(plds_all[wave][1] + ln * 40 + koffA);

    // ---- out += P * V : V B-operands shared by both tiles ----
#pragma unroll
    for (int j = 0; j < 4; ++j) {
      acc[0][j] = WMMA_BF16(pa[0], vb[j], acc[0][j]);
      acc[1][j] = WMMA_BF16(pa[1], vb[j], acc[1][j]);
    }
  }

  // ---- normalize and store fp32 output (B,S,D) ----
#pragma unroll
  for (int t = 0; t < 2; ++t)
#pragma unroll
    for (int r = 0; r < 8; ++r) {
      float inv = 1.0f / lrow[t][r];
      int row = q0 + 16 * t + r + 8 * half;
      float* orow = out + ((size_t)b * SEQ + row) * DMODEL + h * HDIM;
      orow[ln]      = acc[t][0][r] * inv;
      orow[16 + ln] = acc[t][1][r] * inv;
      orow[32 + ln] = acc[t][2][r] * inv;
      orow[48 + ln] = acc[t][3][r] * inv;
    }
}

// ------------------------------------------------------------------
extern "C" void kernel_launch(void* const* d_in, const int* in_sizes, int n_in,
                              void* d_out, int out_size, void* d_ws, size_t ws_size,
                              hipStream_t stream) {
  const float* q = (const float*)d_in[0];
  const float* k = (const float*)d_in[1];
  const float* v = (const float*)d_in[2];
  float* out = (float*)d_out;

  const size_t elems = (size_t)NBH * SEQ * HDIM;        // 4,194,304
  unsigned short* qb = (unsigned short*)d_ws;
  unsigned short* kb = qb + elems;
  unsigned short* vt = kb + elems;

  fa_prep_kernel<<<dim3((unsigned)(elems / 256)), dim3(256), 0, stream>>>(
      q, k, v, qb, kb, vt);

  const unsigned nblk = NBH * (SEQ / 128);              // 512
  fa_attn_kernel<<<dim3(nblk), dim3(128), 0, stream>>>(qb, kb, vt, out);
}
